// LFFModule_33758442946776
// MI455X (gfx1250) — matrix operand order repaired and checked by reading, not compiled, not run on hardware
//
#include <hip/hip_runtime.h>
#include <hip/hip_bf16.h>

typedef _Float16 v16h __attribute__((ext_vector_type(16)));
typedef _Float16 v8h  __attribute__((ext_vector_type(8)));
typedef float    v8f  __attribute__((ext_vector_type(8)));

#define BB   4096
#define VV   20
#define DD   768
#define ROWS (BB * VV)        // 81920
#define WELEM (DD * DD)       // 589824
#define SA   776              // padded stride (halfs): 1552B -> LDS bank step 4
#define COLH 776              // B panel column stride in halfs (1552B)
#define MATH (16 * COLH)      // halfs per weight-matrix panel block (12416)
#define BUFH (2 * MATH)       // halfs per panel buffer (24832 = 49664 B)
#define AS_BYTES (64 * SA * 2)        // 99328
#define BS_BYTES (2 * BUFH * 2)       // 99328
#define FS_BYTES (64 * SA * 2)        // 99328 (fused rows, f16)

// ---------------- Kernel 1: W1/W2 fp32 -> f16 ----------------
__global__ void cvt_weights(const float* __restrict__ W1, const float* __restrict__ W2,
                            _Float16* __restrict__ Wh1, _Float16* __restrict__ Wh2) {
    int i = blockIdx.x * 256 + threadIdx.x;
    if (i < WELEM) {
        Wh1[i] = (_Float16)W1[i];
        Wh2[i] = (_Float16)W2[i];
    }
}

// ---------------- Kernel 2: row softmax (768) -> f16 ----------------
__global__ __launch_bounds__(256) void softmax_rows(const float* __restrict__ vf,
                                                    _Float16* __restrict__ X) {
    __shared__ float red[256];
    const int row = blockIdx.x;
    const float* x = vf + (size_t)row * DD;
    const int t = threadIdx.x;

    float v0 = x[t], v1 = x[t + 256], v2 = x[t + 512];
    float m = fmaxf(v0, fmaxf(v1, v2));
    red[t] = m; __syncthreads();
    #pragma unroll
    for (int s = 128; s > 0; s >>= 1) {
        if (t < s) red[t] = fmaxf(red[t], red[t + s]);
        __syncthreads();
    }
    m = red[0]; __syncthreads();

    float e0 = expf(v0 - m), e1 = expf(v1 - m), e2 = expf(v2 - m);
    red[t] = e0 + e1 + e2; __syncthreads();
    #pragma unroll
    for (int s = 128; s > 0; s >>= 1) {
        if (t < s) red[t] += red[t + s];
        __syncthreads();
    }
    float inv = 1.0f / red[0];

    _Float16* o = X + (size_t)row * DD;
    o[t]       = (_Float16)(e0 * inv);
    o[t + 256] = (_Float16)(e1 * inv);
    o[t + 512] = (_Float16)(e2 * inv);
}

// ---- CDNA5 async global->LDS copy (ASYNCcnt), 16B per lane ----
__device__ __forceinline__ void async_copy_b128(uint32_t lds_off, const _Float16* gsrc) {
    asm volatile("global_load_async_to_lds_b128 %0, %1, off"
                 :: "v"(lds_off), "v"((uint64_t)(uintptr_t)gsrc)
                 : "memory");
}
__device__ __forceinline__ void wait_async0() {
    asm volatile("s_wait_asynccnt 0x0" ::: "memory");
}

// Issue one 48KB B-panel (both matrices, 16 columns x 768 k, f16) into LDS.
// 3072 x 16B chunks over 128 threads -> 24 async b128 per thread.
__device__ __forceinline__ void issue_panel(const _Float16* __restrict__ Wh1,
                                            const _Float16* __restrict__ Wh2,
                                            int n0, uint32_t bs_base, int tid) {
    #pragma unroll
    for (int t = 0; t < 24; ++t) {
        int idx  = tid + t * 128;
        int pair = idx / 96;          // (mat,col) 0..31
        int ch   = idx % 96;          // 16B chunk within the 1536B column
        int mat  = pair >> 4;
        int col  = pair & 15;
        const _Float16* src = (mat ? Wh2 : Wh1) + (size_t)(n0 + col) * DD + ch * 8;
        uint32_t dst = bs_base + (uint32_t)(mat * (MATH * 2) + col * (COLH * 2) + ch * 16);
        async_copy_b128(dst, src);
    }
}

// ---------------- Kernel 3: dual WMMA GEMM + fuse + 2x conv1d(k=3) ----------------
__global__ __launch_bounds__(128) void lff_main(
    const _Float16* __restrict__ X, const _Float16* __restrict__ Wh1,
    const _Float16* __restrict__ Wh2, const float* __restrict__ text,
    const float* __restrict__ b1, const float* __restrict__ b2,
    const float* __restrict__ cw1, const float* __restrict__ cb1,
    const float* __restrict__ cw2, const float* __restrict__ cb2,
    float* __restrict__ out) {

    extern __shared__ char smem[];
    _Float16* As = (_Float16*)smem;                          // 64 x SA halfs
    _Float16* Bs = (_Float16*)(smem + AS_BYTES);             // 2 x BUFH halfs
    _Float16* Fs = (_Float16*)(smem + AS_BYTES + BS_BYTES);  // 64 x SA halfs

    const int tid  = threadIdx.x;
    const int R0   = blockIdx.x * 64;
    const int wave = tid >> 5;
    const int lane = tid & 31;
    const int lr   = lane & 15;     // column / A-row selector
    const int kh   = lane >> 4;     // K half selector

    // LDS byte offset of the B panel region (flat-address low 32 bits == LDS offset)
    const uint32_t bsBase = (uint32_t)(uintptr_t)(void*)Bs;

    // Kick off async copy of the first B panel while we stage A rows.
    issue_panel(Wh1, Wh2, 0, bsBase, tid);

    // ---- Stage 64 activation rows (f16) into LDS, 16B chunks, coalesced ----
    for (int idx = tid; idx < 64 * 48; idx += 128) {
        int row = idx / 48, ch = idx % 48;
        const uint4* src = (const uint4*)(X + (size_t)(R0 + row) * DD + ch * 8);
        *(uint4*)(As + row * SA + ch * 8) = *src;
    }
    wait_async0();
    __syncthreads();

    // ---- GEMM: each wave computes 16 rows x 768 cols, both weight matrices ----
    const _Float16* ap = As + (wave * 16 + lr) * SA + kh * 16;

    int buf = 0;
    for (int nt = 0; nt < 48; ++nt) {
        // Prefetch next panel into the other buffer (overlaps with the WMMAs below).
        if (nt < 47)
            issue_panel(Wh1, Wh2, (nt + 1) * 16, bsBase + (uint32_t)((buf ^ 1) * (BUFH * 2)), tid);

        const int n = nt * 16 + lr;
        const float b1v = b1[n], b2v = b2[n];

        v8f acc1, acc2;
        #pragma unroll
        for (int i = 0; i < 8; ++i) { acc1[i] = b1v; acc2[i] = b2v; }

        const _Float16* bp1 = Bs + buf * BUFH + lr * COLH + kh * 16;
        const _Float16* bp2 = bp1 + MATH;

        // Software-pipelined k loop: fragments for kt+1 loaded before WMMAs of kt.
        v8h alo = *(const v8h*)(ap);
        v8h ahi = *(const v8h*)(ap + 8);
        v16h a_c  = __builtin_shufflevector(alo, ahi, 0,1,2,3,4,5,6,7,8,9,10,11,12,13,14,15);
        v16h b1_c = *(const v16h*)(bp1);
        v16h b2_c = *(const v16h*)(bp2);

        #pragma unroll
        for (int kt = 0; kt < 24; ++kt) {
            v16h a_n = a_c, b1_n = b1_c, b2_n = b2_c;
            if (kt < 23) {
                v8h nlo = *(const v8h*)(ap + (kt + 1) * 32);
                v8h nhi = *(const v8h*)(ap + (kt + 1) * 32 + 8);
                a_n  = __builtin_shufflevector(nlo, nhi, 0,1,2,3,4,5,6,7,8,9,10,11,12,13,14,15);
                b1_n = *(const v16h*)(bp1 + (kt + 1) * 32);
                b2_n = *(const v16h*)(bp2 + (kt + 1) * 32);
            }
            acc1 = __builtin_amdgcn_wmma_f32_16x16x32_f16(false, a_c, false, b1_c,
                                                          (short)0, acc1, false, false);
            acc2 = __builtin_amdgcn_wmma_f32_16x16x32_f16(false, a_c, false, b2_c,
                                                          (short)0, acc2, false, false);
            a_c = a_n; b1_c = b1_n; b2_c = b2_n;
        }

        // Epilogue: fused = relu(text * m1 + m2) -> LDS (f16)
        #pragma unroll
        for (int r = 0; r < 8; ++r) {
            const int ml = wave * 16 + kh * 8 + r;   // local row (D-matrix layout)
            const int rg = R0 + ml;                  // global (b*V+v) row
            const float t = text[(size_t)(rg / VV) * DD + n];
            Fs[ml * SA + n] = (_Float16)fmaxf(t * acc1[r] + acc2[r], 0.0f);
        }

        // Next panel must be resident (and this buffer free) before next iteration.
        wait_async0();
        __syncthreads();
        buf ^= 1;
    }

    // ---- Two conv1d(k=3, pad=1) along D, relu between, write output ----
    const float w10 = cw1[0], w11 = cw1[1], w12 = cw1[2], q1 = cb1[0];
    const float w20 = cw2[0], w21 = cw2[1], w22 = cw2[2], q2 = cb2[0];

    for (int r = 0; r < 16; ++r) {
        const int ml = wave * 16 + r;
        const _Float16* Frow = Fs + ml * SA;
        const int c0 = lane * 24;                 // 32 lanes x 24 cols = 768

        float x[28];
        #pragma unroll
        for (int i = 0; i < 28; ++i) {
            int col = c0 - 2 + i;
            x[i] = (col >= 0 && col < DD) ? (float)Frow[col] : 0.0f;
        }
        float y[26];
        #pragma unroll
        for (int j = 0; j < 26; ++j) {
            int col = c0 - 1 + j;
            float v = w10 * x[j] + w11 * x[j + 1] + w12 * x[j + 2] + q1;
            y[j] = (col >= 0 && col < DD) ? fmaxf(v, 0.0f) : 0.0f;
        }
        const size_t ob = (size_t)(R0 + ml) * DD + c0;
        #pragma unroll
        for (int i = 0; i < 24; i += 4) {
            float4 o4;
            o4.x = w20 * y[i]     + w21 * y[i + 1] + w22 * y[i + 2] + q2;
            o4.y = w20 * y[i + 1] + w21 * y[i + 2] + w22 * y[i + 3] + q2;
            o4.z = w20 * y[i + 2] + w21 * y[i + 3] + w22 * y[i + 4] + q2;
            o4.w = w20 * y[i + 3] + w21 * y[i + 4] + w22 * y[i + 5] + q2;
            *(float4*)(out + ob + i) = o4;
        }
    }
}

extern "C" void kernel_launch(void* const* d_in, const int* in_sizes, int n_in,
                              void* d_out, int out_size, void* d_ws, size_t ws_size,
                              hipStream_t stream) {
    const float* text = (const float*)d_in[0];
    const float* vf   = (const float*)d_in[1];
    const float* W1   = (const float*)d_in[2];
    const float* b1   = (const float*)d_in[3];
    const float* W2   = (const float*)d_in[4];
    const float* b2   = (const float*)d_in[5];
    const float* cw1  = (const float*)d_in[6];
    const float* cb1  = (const float*)d_in[7];
    const float* cw2  = (const float*)d_in[8];
    const float* cb2  = (const float*)d_in[9];
    float* out = (float*)d_out;

    // workspace: Wh1 | Wh2 | X  (all f16)
    _Float16* Wh1 = (_Float16*)d_ws;
    _Float16* Wh2 = Wh1 + WELEM;
    _Float16* X   = Wh2 + WELEM;

    cvt_weights<<<(WELEM + 255) / 256, 256, 0, stream>>>(W1, W2, Wh1, Wh2);
    softmax_rows<<<ROWS, 256, 0, stream>>>(vf, X);

    size_t smem = (size_t)AS_BYTES + BS_BYTES + FS_BYTES;   // 297,984 B < 320 KB
    lff_main<<<ROWS / 64, 128, smem, stream>>>(X, Wh1, Wh2, text, b1, b2,
                                               cw1, cb1, cw2, cb2, out);
}
